// PointNet_SA_Module_MSG_2430951489688
// MI455X (gfx1250) — compile-verified, matrix operand order checked by reading, not run on hardware
//
#include <hip/hip_runtime.h>

#define B_ 8
#define N_ 16384
#define M_ 1024
#define EPS_ 1e-5f

typedef unsigned int u32;
typedef __bf16 v16bf __attribute__((ext_vector_type(16)));
typedef float v8f __attribute__((ext_vector_type(8)));
typedef unsigned short v8us __attribute__((ext_vector_type(8)));

__device__ inline unsigned short f2bf(float f) {
  unsigned x = __float_as_uint(f);
  unsigned r = x + 0x7FFFu + ((x >> 16) & 1u);
  return (unsigned short)(r >> 16);
}
__device__ inline float bf2f(unsigned short u) {
  return __uint_as_float(((unsigned)u) << 16);
}
__device__ inline __bf16 usbf(unsigned short u) {
  __bf16 b; __builtin_memcpy(&b, &u, 2); return b;
}

// ---------------------------------------------------------------------------
// 1) Farthest point sampling: 1 block per batch, 1024 threads, 16 pts/thread
//    in registers. Writes new_xyz to d_out (first 8*1024*3 floats) and ws.
// ---------------------------------------------------------------------------
__global__ __launch_bounds__(1024)
void fps_kernel(const float* __restrict__ xyz, float* __restrict__ nx_out,
                float* __restrict__ nx_ws) {
  const int b = blockIdx.x;
  const int t = threadIdx.x;
  const float* bx = xyz + (size_t)b * N_ * 3;
  float px[16], py[16], pz[16], dd[16];
#pragma unroll
  for (int i = 0; i < 16; i++) {
    int n = i * 1024 + t;
    px[i] = bx[n * 3 + 0]; py[i] = bx[n * 3 + 1]; pz[i] = bx[n * 3 + 2];
    dd[i] = 1e10f;
  }
  __shared__ float sc[3];
  __shared__ float sBest[32];
  __shared__ int   sBestN[32];
  __shared__ int   sNext;
  if (t == 0) { sc[0] = px[0]; sc[1] = py[0]; sc[2] = pz[0]; sNext = 0; }
  __syncthreads();
  const int lane = t & 31, wv = t >> 5;
  for (int m = 0; m < M_; m++) {
    float cx = sc[0], cy = sc[1], cz = sc[2];
    if (t == 0) {
      float* o = nx_out + ((size_t)b * M_ + m) * 3;
      o[0] = cx; o[1] = cy; o[2] = cz;
      float* w = nx_ws + ((size_t)b * M_ + m) * 3;
      w[0] = cx; w[1] = cy; w[2] = cz;
    }
    float best = -1.0f; int bestN = 0x7fffffff;
#pragma unroll
    for (int i = 0; i < 16; i++) {
      float dx = __fsub_rn(px[i], cx);
      float dy = __fsub_rn(py[i], cy);
      float dz = __fsub_rn(pz[i], cz);
      float d = __fadd_rn(__fadd_rn(__fmul_rn(dx, dx), __fmul_rn(dy, dy)),
                          __fmul_rn(dz, dz));
      dd[i] = fminf(dd[i], d);
      int n = i * 1024 + t;
      if (dd[i] > best || (dd[i] == best && n < bestN)) { best = dd[i]; bestN = n; }
    }
#pragma unroll
    for (int off = 16; off > 0; off >>= 1) {
      float ob = __shfl_xor(best, off, 32);
      int   on = __shfl_xor(bestN, off, 32);
      if (ob > best || (ob == best && on < bestN)) { best = ob; bestN = on; }
    }
    if (lane == 0) { sBest[wv] = best; sBestN[wv] = bestN; }
    __syncthreads();
    if (wv == 0) {
      float bb = sBest[lane]; int bn = sBestN[lane];
#pragma unroll
      for (int off = 16; off > 0; off >>= 1) {
        float ob = __shfl_xor(bb, off, 32);
        int   on = __shfl_xor(bn, off, 32);
        if (ob > bb || (ob == bb && on < bn)) { bb = ob; bn = on; }
      }
      if (lane == 0) sNext = bn;
    }
    __syncthreads();
    int nx = sNext;
    if ((nx & 1023) == t) { int i = nx >> 10; sc[0] = px[i]; sc[1] = py[i]; sc[2] = pz[i]; }
    __syncthreads();
  }
}

// ---------------------------------------------------------------------------
// 2) Fused 3-scale ball query: one wave per (b,m). Ascending-index append
//    via ballot+prefix (== reference's masked-sort semantics), pad with first.
// ---------------------------------------------------------------------------
__global__ __launch_bounds__(256)
void ballquery_kernel(const float* __restrict__ xyz, const float* __restrict__ nx,
                      int* __restrict__ i0, int* __restrict__ i1, int* __restrict__ i2) {
  const int gw = (int)((blockIdx.x * blockDim.x + threadIdx.x) >> 5);
  const int lane = threadIdx.x & 31;
  const int b = gw / M_;
  const float* c = nx + (size_t)gw * 3;
  const float cx = c[0], cy = c[1], cz = c[2];
  const float* bx = xyz + (size_t)b * N_ * 3;
  int* o0 = i0 + (size_t)gw * 16;
  int* o1 = i1 + (size_t)gw * 32;
  int* o2 = i2 + (size_t)gw * 128;
  int cnt0 = 0, cnt1 = 0, cnt2 = 0;
  int f0 = 0, f1 = 0, f2 = 0;
  const float r20 = 0.1f * 0.1f, r21 = 0.2f * 0.2f, r22 = 0.4f * 0.4f;
  const unsigned below = (1u << lane) - 1u;
  for (int n0 = 0; n0 < N_; n0 += 32) {
    int n = n0 + lane;
    float dx = __fsub_rn(bx[n * 3 + 0], cx);
    float dy = __fsub_rn(bx[n * 3 + 1], cy);
    float dz = __fsub_rn(bx[n * 3 + 2], cz);
    float d2 = __fadd_rn(__fadd_rn(__fmul_rn(dx, dx), __fmul_rn(dy, dy)),
                         __fmul_rn(dz, dz));
    if (cnt0 < 16) {
      bool in = d2 <= r20;
      unsigned mk = (unsigned)__ballot(in);
      if (cnt0 == 0 && mk) f0 = n0 + __ffs(mk) - 1;
      int pos = cnt0 + __popc(mk & below);
      if (in && pos < 16) o0[pos] = n;
      cnt0 += __popc(mk);
    }
    if (cnt1 < 32) {
      bool in = d2 <= r21;
      unsigned mk = (unsigned)__ballot(in);
      if (cnt1 == 0 && mk) f1 = n0 + __ffs(mk) - 1;
      int pos = cnt1 + __popc(mk & below);
      if (in && pos < 32) o1[pos] = n;
      cnt1 += __popc(mk);
    }
    if (cnt2 < 128) {
      bool in = d2 <= r22;
      unsigned mk = (unsigned)__ballot(in);
      if (cnt2 == 0 && mk) f2 = n0 + __ffs(mk) - 1;
      int pos = cnt2 + __popc(mk & below);
      if (in && pos < 128) o2[pos] = n;
      cnt2 += __popc(mk);
    }
    if (cnt0 >= 16 && cnt1 >= 32 && cnt2 >= 128) break;
  }
  for (int p = min(cnt0, 16) + lane; p < 16; p += 32) o0[p] = f0;
  for (int p = min(cnt1, 32) + lane; p < 32; p += 32) o1[p] = f1;
  for (int p = min(cnt2, 128) + lane; p < 128; p += 32) o2[p] = f2;
}

// ---------------------------------------------------------------------------
// 3) Gather + concat -> bf16 feat0 rows, channel-padded 67 -> 96
// ---------------------------------------------------------------------------
template <int K>
__global__ __launch_bounds__(256)
void gather_kernel(const float* __restrict__ xyz, const float* __restrict__ pts,
                   const float* __restrict__ nx, const int* __restrict__ idx,
                   unsigned short* __restrict__ feat) {
  const int row = blockIdx.x * 256 + threadIdx.x;
  const int g = idx[row];
  const int bm = row / K;
  const int b = bm / M_;
  const float* p = xyz + ((size_t)b * N_ + g) * 3;
  const float* c = nx + (size_t)bm * 3;
  float vals[96];
  vals[0] = __fsub_rn(p[0], c[0]);
  vals[1] = __fsub_rn(p[1], c[1]);
  vals[2] = __fsub_rn(p[2], c[2]);
  const float4* pr = (const float4*)(pts + ((size_t)b * N_ + g) * 64);
  float4 v[16];
#pragma unroll
  for (int i = 0; i < 16; i++) v[i] = pr[i];
#pragma unroll
  for (int i = 0; i < 16; i++) {
    vals[3 + 4 * i] = v[i].x; vals[4 + 4 * i] = v[i].y;
    vals[5 + 4 * i] = v[i].z; vals[6 + 4 * i] = v[i].w;
  }
#pragma unroll
  for (int e = 67; e < 96; e++) vals[e] = 0.0f;
  u32* o32 = (u32*)(feat + (size_t)row * 96);
#pragma unroll
  for (int j = 0; j < 48; j++)
    o32[j] = (u32)f2bf(vals[2 * j]) | ((u32)f2bf(vals[2 * j + 1]) << 16);
}

// ---------------------------------------------------------------------------
// 4) Weight prep: f32 W[cin][cout] -> bf16 Wt[COUT][CIN] (zero-padded K)
// ---------------------------------------------------------------------------
__global__ __launch_bounds__(256)
void prepw_kernel(const float* __restrict__ W, unsigned short* __restrict__ Wt,
                  int cin, int CIN, int COUT) {
  int i = blockIdx.x * 256 + threadIdx.x;
  if (i >= COUT * CIN) return;
  int c = i / CIN, k = i % CIN;
  float v = (k < cin) ? W[(size_t)k * COUT + c] : 0.0f;
  Wt[i] = f2bf(v);
}

__global__ void zerostats_kernel(float* s) { s[threadIdx.x] = 0.0f; }

// ---------------------------------------------------------------------------
// 5) WMMA GEMM: 1 wave = 16 rows x COUT. Optional fused (affine+ReLU) on A
//    load (= batchnorm of the previous layer). bf16 in, f32 acc, bf16 out.
// ---------------------------------------------------------------------------
template <int CIN, int COUT>
__global__ __launch_bounds__(256)
void gemm_kernel(const unsigned short* __restrict__ A,
                 const unsigned short* __restrict__ Wt,
                 const float* __restrict__ affine,
                 unsigned short* __restrict__ Y, int useAffine) {
  __shared__ unsigned short sW[COUT * CIN];
  __shared__ float sAf[2 * CIN];
  for (int i = threadIdx.x; i < COUT * CIN / 2; i += 256)
    ((u32*)sW)[i] = ((const u32*)Wt)[i];
  if (useAffine)
    for (int i = threadIdx.x; i < 2 * CIN; i += 256) sAf[i] = affine[i];
  __syncthreads();
  const int lane = threadIdx.x & 31, wv = threadIdx.x >> 5;
  const long row16 = ((long)blockIdx.x * 8 + wv) * 16;
  const int r = lane & 15, hi = lane >> 4;
  const unsigned short* arow = A + (size_t)(row16 + r) * CIN;
  v8f acc[COUT / 16];
#pragma unroll
  for (int cb = 0; cb < COUT / 16; cb++)
#pragma unroll
    for (int j = 0; j < 8; j++) acc[cb][j] = 0.0f;
#pragma unroll
  for (int ks = 0; ks < CIN / 32; ks++) {
    const int kb = ks * 32;
    v8us lo = *(const v8us*)(arow + kb + hi * 8);
    v8us hh = *(const v8us*)(arow + kb + 16 + hi * 8);
    v16bf a;
    if (useAffine) {
#pragma unroll
      for (int e = 0; e < 8; e++) {
        int c0 = kb + hi * 8 + e;
        float g0 = fmaxf(bf2f(lo[e]) * sAf[2 * c0] + sAf[2 * c0 + 1], 0.0f);
        a[e] = usbf(f2bf(g0));
        int c1 = kb + 16 + hi * 8 + e;
        float g1 = fmaxf(bf2f(hh[e]) * sAf[2 * c1] + sAf[2 * c1 + 1], 0.0f);
        a[e + 8] = usbf(f2bf(g1));
      }
    } else {
#pragma unroll
      for (int e = 0; e < 8; e++) { a[e] = usbf(lo[e]); a[e + 8] = usbf(hh[e]); }
    }
#pragma unroll
    for (int cb = 0; cb < COUT / 16; cb++) {
      const unsigned short* wrow = sW + (cb * 16 + r) * CIN;
      v8us bl = *(const v8us*)(wrow + kb + hi * 8);
      v8us bh = *(const v8us*)(wrow + kb + 16 + hi * 8);
      v16bf bb;
#pragma unroll
      for (int e = 0; e < 8; e++) { bb[e] = usbf(bl[e]); bb[e + 8] = usbf(bh[e]); }
      acc[cb] = __builtin_amdgcn_wmma_f32_16x16x32_bf16(
          false, a, false, bb, (short)0, acc[cb], false, false);
    }
  }
#pragma unroll
  for (int cb = 0; cb < COUT / 16; cb++)
#pragma unroll
    for (int vg = 0; vg < 8; vg++) {
      long ro = row16 + hi * 8 + vg;
      Y[(size_t)ro * COUT + cb * 16 + r] = f2bf(acc[cb][vg]);
    }
}

// ---------------------------------------------------------------------------
// 6) Per-channel sum/sumsq (LDS atomics, lane-rotated channels), then global
// ---------------------------------------------------------------------------
__global__ __launch_bounds__(256)
void stats_kernel(const unsigned short* __restrict__ Y, float* __restrict__ st,
                  long R, int COUT) {
  __shared__ float ls[256];
  for (int i = threadIdx.x; i < 2 * COUT; i += 256) ls[i] = 0.0f;
  __syncthreads();
  const int lane = threadIdx.x & 31;
  const long stride = (long)gridDim.x * blockDim.x;
  for (long row = (long)blockIdx.x * blockDim.x + threadIdx.x; row < R; row += stride) {
    const unsigned short* yr = Y + (size_t)row * COUT;
    for (int cc = 0; cc < COUT; cc++) {
      int c = cc + lane; if (c >= COUT) c -= COUT;
      float v = bf2f(yr[c]);
      atomicAdd(&ls[2 * c], v);
      atomicAdd(&ls[2 * c + 1], v * v);
    }
  }
  __syncthreads();
  for (int i = threadIdx.x; i < 2 * COUT; i += 256) atomicAdd(&st[i], ls[i]);
}

__global__ void finalize_kernel(const float* st, const float* gamma,
                                const float* beta, float* affine,
                                float invcount, int COUT) {
  int c = threadIdx.x;
  if (c < COUT) {
    float mu = st[2 * c] * invcount;
    float var = st[2 * c + 1] * invcount - mu * mu;
    float sc = gamma[c] * rsqrtf(var + EPS_);
    affine[2 * c] = sc;
    affine[2 * c + 1] = beta[c] - mu * sc;
  }
}

// ---------------------------------------------------------------------------
// 7) Fused affine+ReLU + max over K, write into d_out feature region
// ---------------------------------------------------------------------------
__global__ void maxpool_kernel(const unsigned short* __restrict__ Y,
                               const float* __restrict__ affine,
                               float* __restrict__ out, int K, int COUT,
                               int outOff) {
  const int bm = blockIdx.x, c = threadIdx.x;
  const float sc = affine[2 * c], sh = affine[2 * c + 1];
  const unsigned short* yr = Y + (size_t)bm * K * COUT + c;
  float mx = 0.0f;
  for (int kk = 0; kk < K; kk++)
    mx = fmaxf(mx, fmaxf(bf2f(yr[(size_t)kk * COUT]) * sc + sh, 0.0f));
  out[(size_t)bm * 320 + outOff + c] = mx;
}

// ---------------------------------------------------------------------------
static void launch_gemm(int cin, int cout, const unsigned short* A,
                        const unsigned short* Wt, const float* af,
                        unsigned short* Y, long R, int useAf, hipStream_t s) {
  dim3 g((unsigned)(R / 128)), b(256);
#define GCASE(CI, CO)                                                        \
  if (cin == CI && cout == CO) {                                             \
    gemm_kernel<CI, CO><<<g, b, 0, s>>>(A, Wt, af, Y, useAf);                \
    return;                                                                  \
  }
  GCASE(96, 32) GCASE(32, 32) GCASE(32, 64)
  GCASE(96, 64) GCASE(64, 64) GCASE(64, 128)
  GCASE(64, 96) GCASE(96, 128)
#undef GCASE
}

extern "C" void kernel_launch(void* const* d_in, const int* in_sizes, int n_in,
                              void* d_out, int out_size, void* d_ws, size_t ws_size,
                              hipStream_t stream) {
  (void)in_sizes; (void)n_in; (void)out_size; (void)ws_size;
  const float* xyz = (const float*)d_in[0];
  const float* pts = (const float*)d_in[1];
  float* out = (float*)d_out;

  char* w = (char*)d_ws;
  size_t off = 0;
  auto alloc = [&](size_t bytes) {
    size_t p = off;
    off += (bytes + 255) & ~(size_t)255;
    return p;
  };
  float* nx_ws = (float*)(w + alloc((size_t)B_ * M_ * 3 * 4));
  int* idx0 = (int*)(w + alloc((size_t)B_ * M_ * 16 * 4));
  int* idx1 = (int*)(w + alloc((size_t)B_ * M_ * 32 * 4));
  int* idx2 = (int*)(w + alloc((size_t)B_ * M_ * 128 * 4));
  float* stats = (float*)(w + alloc(256 * 4));
  float* affine = (float*)(w + alloc(256 * 4));
  unsigned short* wt = (unsigned short*)(w + alloc((size_t)128 * 96 * 2));
  unsigned short* bufA = (unsigned short*)(w + alloc((size_t)B_ * M_ * 128 * 128 * 2));
  unsigned short* bufB = (unsigned short*)(w + alloc((size_t)B_ * M_ * 128 * 128 * 2));

  fps_kernel<<<B_, 1024, 0, stream>>>(xyz, out, nx_ws);
  ballquery_kernel<<<(B_ * M_) / 8, 256, 0, stream>>>(xyz, nx_ws, idx0, idx1, idx2);

  static const int KSC[3] = {16, 32, 128};
  static const int LCIN[3][3] = {{96, 32, 32}, {96, 64, 64}, {96, 64, 96}};
  static const int LCOUT[3][3] = {{32, 32, 64}, {64, 64, 128}, {64, 96, 128}};
  static const int OUTOFF[3] = {0, 64, 192};
  float* outFeat = out + (size_t)B_ * M_ * 3;

  for (int s = 0; s < 3; s++) {
    const int K = KSC[s];
    const long R = (long)B_ * M_ * K;
    if (s == 0) gather_kernel<16><<<(unsigned)(R / 256), 256, 0, stream>>>(xyz, pts, nx_ws, idx0, bufA);
    if (s == 1) gather_kernel<32><<<(unsigned)(R / 256), 256, 0, stream>>>(xyz, pts, nx_ws, idx1, bufA);
    if (s == 2) gather_kernel<128><<<(unsigned)(R / 256), 256, 0, stream>>>(xyz, pts, nx_ws, idx2, bufA);

    unsigned short* cur = bufA;
    unsigned short* nxt = bufB;
    for (int l = 0; l < 3; l++) {
      const float* W = (const float*)d_in[2 + s * 9 + l * 3];
      const float* gamma = (const float*)d_in[2 + s * 9 + l * 3 + 1];
      const float* beta = (const float*)d_in[2 + s * 9 + l * 3 + 2];
      const int cin = (l == 0) ? 67 : LCOUT[s][l - 1];
      const int CIN = LCIN[s][l], COUT = LCOUT[s][l];
      prepw_kernel<<<(COUT * CIN + 255) / 256, 256, 0, stream>>>(W, wt, cin, CIN, COUT);
      zerostats_kernel<<<1, 256, 0, stream>>>(stats);
      launch_gemm(CIN, COUT, cur, wt, affine, nxt, R, l > 0 ? 1 : 0, stream);
      stats_kernel<<<512, 256, 0, stream>>>(nxt, stats, R, COUT);
      finalize_kernel<<<1, 128, 0, stream>>>(stats, gamma, beta, affine,
                                             1.0f / (float)R, COUT);
      unsigned short* t = cur; cur = nxt; nxt = t;
    }
    maxpool_kernel<<<B_ * M_, LCOUT[s][2], 0, stream>>>(cur, affine, outFeat, K,
                                                        LCOUT[s][2], OUTOFF[s]);
  }
}
